// QuantLinear_46909632807707
// MI455X (gfx1250) — matrix operand order repaired and checked by reading, not compiled, not run on hardware
//
#include <hip/hip_runtime.h>

// ---------------------------------------------------------------------------
// QuIP#-style quantized linear for MI455X (gfx1250, wave32, WMMA).
//
// y = fwht( fwht(x*SU) @ (grid[Qidxs])^T ) * SV + bias
//
// GEMM (M=8192,N=4096,K=4096, ~275 GFLOP) dominates -> bf16 WMMA
// (v_wmma_f32_16x16x32_bf16), fp32 accumulate. bf16 weights (32MB) +
// bf16 activations (64MB) fit in 192MB L2 -> compute bound.
//
// Double-buffered LDS + CDNA5 async global->LDS staging
// (GLOBAL_LOAD_ASYNC_TO_LDS_B128, ASYNCcnt): tile loads overlap WMMA, one
// barrier per K-step. K-loop at unroll 1 with a 32-bit buffer-offset toggle
// (stable accumulator registers); fragment reads as explicit 16B LDS
// vector loads from alignas(16) buffers -> ds_load_b128.
// ---------------------------------------------------------------------------

typedef __attribute__((ext_vector_type(16))) __bf16 v16bf;
typedef __attribute__((ext_vector_type(8)))  float  v8f;

#define IN_F   4096
#define OUT_F  4096
#define NROWS  8192      // 4*2048 flattened batch rows
#define THREADS 256

// ------------------------- async copy helpers ------------------------------
#if defined(__has_builtin)
#  if __has_builtin(__builtin_amdgcn_global_load_async_to_lds_b128)
#    define ASYNC_VIA_BUILTIN 1
#  endif
#  if __has_builtin(__builtin_amdgcn_s_wait_asynccnt)
#    define WAIT_VIA_BUILTIN 1
#  endif
#endif

// Exact param type per hipcc diagnostic: 'vector_size(4*sizeof(int)) int AS1*'
typedef int v4i_vs __attribute__((vector_size(16)));
typedef __attribute__((address_space(1))) v4i_vs* as1_v4i_p;
typedef __attribute__((address_space(3))) v4i_vs* as3_v4i_p;

__device__ __forceinline__ void async_copy16(const void* g, void* lds) {
#if ASYNC_VIA_BUILTIN
    __builtin_amdgcn_global_load_async_to_lds_b128(
        (as1_v4i_p)g, (as3_v4i_p)lds, 0, 0);
#else
    unsigned int l = (unsigned int)(unsigned long long)lds;
    asm volatile("global_load_async_to_lds_b128 %0, %1, off"
                 :: "v"(l), "v"(g) : "memory");
#endif
}

__device__ __forceinline__ void wait_async0() {
#if WAIT_VIA_BUILTIN
    __builtin_amdgcn_s_wait_asynccnt(0);
#else
    asm volatile("s_wait_asynccnt 0x0" ::: "memory");
#endif
}

// ---------------- FWHT (normalized) of one 4096-row in LDS ----------------
__device__ __forceinline__ void fwht4096(float* buf, int tid) {
    for (int d = 1; d < IN_F; d <<= 1) {
        #pragma unroll 1
        for (int t = 0; t < 8; ++t) {
            int b   = tid + t * THREADS;          // butterfly id 0..2047
            int off = b & (d - 1);
            int i   = ((b - off) << 1) + off;     // group*2d + off
            float a0 = buf[i];
            float b0 = buf[i + d];
            buf[i]     = a0 + b0;
            buf[i + d] = a0 - b0;
        }
        __syncthreads();
    }
}

// ---------------- Kernel 1: h = fwht(x*SU) -> bf16 ------------------------
__global__ void __launch_bounds__(THREADS)
k_had_in(const float* __restrict__ x, const float* __restrict__ SU,
         __bf16* __restrict__ hbuf) {
    __shared__ float buf[IN_F];
    const int row = blockIdx.x;
    const int tid = threadIdx.x;
    const float* xr = x + (size_t)row * IN_F;
    #pragma unroll
    for (int t = 0; t < 16; ++t) {
        int i = tid + t * THREADS;
        buf[i] = xr[i] * SU[i];
    }
    __syncthreads();
    fwht4096(buf, tid);
    const float inv = 1.0f / 64.0f;               // 1/sqrt(4096)
    __bf16* hr = hbuf + (size_t)row * IN_F;
    #pragma unroll
    for (int t = 0; t < 16; ++t) {
        int i = tid + t * THREADS;
        hr[i] = (__bf16)(buf[i] * inv);
    }
}

// ---------------- Kernel 2: decode codebook -> bf16 W (4096x4096) ---------
__global__ void __launch_bounds__(THREADS)
k_decode(const int* __restrict__ Qidxs, const float* __restrict__ cbk,
         __bf16* __restrict__ W) {
    int t = blockIdx.x * THREADS + threadIdx.x;   // 4096*512 entries
    int idx = Qidxs[t];
    const float4* g = (const float4*)(cbk + (size_t)idx * 8);
    float4 lo = g[0], hi = g[1];
    __bf16* w = W + (size_t)t * 8;
    w[0] = (__bf16)lo.x; w[1] = (__bf16)lo.y;
    w[2] = (__bf16)lo.z; w[3] = (__bf16)lo.w;
    w[4] = (__bf16)hi.x; w[5] = (__bf16)hi.y;
    w[6] = (__bf16)hi.z; w[7] = (__bf16)hi.w;
}

// ---------------- Kernel 3: y = h @ W^T  (bf16 WMMA, fp32 acc) ------------
// Block tile 128x128, BK=32. 8 waves in 4(M)x2(N) grid, each wave owns a
// 32x64 = 2x4 grid of 16x16 WMMA tiles. LDS double buffered; async staging.
#define BM 128
#define BN 128
#define BK 32
#define LDP 40   // LDS row pitch (bf16): 80B rows, 16B-aligned chunks,
                 // 20*m mod 64 banks distinct over 16 rows -> conflict-clean
#define BUFSZ (BM * LDP)   // elements per buffer (A and B identical shape)

union FragBF { v16bf v; v4i_vs q[2]; unsigned int u[8]; };

__global__ void __launch_bounds__(THREADS)
k_gemm(const __bf16* __restrict__ A,   // 8192 x 4096 (row-major, M x K)
       const __bf16* __restrict__ B,   // 4096 x 4096 (row-major, N x K)  == W
       float* __restrict__ C) {        // 8192 x 4096
    __shared__ alignas(16) __bf16 AshF[2 * BUFSZ];
    __shared__ alignas(16) __bf16 BshF[2 * BUFSZ];

    const int tid  = threadIdx.x;
    const int wave = tid >> 5;
    const int lane = tid & 31;
    const int m16  = lane & 15;
    const int half = lane >> 4;
    const int waveM = wave >> 1;          // 0..3 -> +32 rows each
    const int waveN = wave & 1;           // 0..1 -> +64 cols each
    const int mBase0 = blockIdx.y * BM;
    const int nBase0 = blockIdx.x * BN;

    // staging: 512 x 16B per tile per matrix; thread -> quad q=tid&3, rows r0,+64
    const int q  = tid & 3;               // k chunk: q*8 bf16 (16B)
    const int r0 = tid >> 2;              // 0..63
    const unsigned stA0 = (unsigned)((r0      ) * LDP + q * 8);
    const unsigned stA1 = (unsigned)((r0 + 64 ) * LDP + q * 8);

    // per-thread fragment row bases (element offsets, constant over K loop)
    unsigned aRow[2], bRow[4];
    #pragma unroll
    for (int i = 0; i < 2; ++i)
        aRow[i] = (unsigned)((waveM * 32 + i * 16 + m16) * LDP);
    #pragma unroll
    for (int i = 0; i < 4; ++i)
        bRow[i] = (unsigned)((waveN * 64 + i * 16 + m16) * LDP);

    v8f acc[2][4];
    #pragma unroll
    for (int i = 0; i < 2; ++i)
        #pragma unroll
        for (int j = 0; j < 4; ++j)
            acc[i][j] = {};

    // ---- prologue: stage tile kb=0 into buffer 0 ----
    async_copy16(A + (size_t)(mBase0 + r0     ) * IN_F + q * 8, &AshF[stA0]);
    async_copy16(A + (size_t)(mBase0 + r0 + 64) * IN_F + q * 8, &AshF[stA1]);
    async_copy16(B + (size_t)(nBase0 + r0     ) * IN_F + q * 8, &BshF[stA0]);
    async_copy16(B + (size_t)(nBase0 + r0 + 64) * IN_F + q * 8, &BshF[stA1]);
    wait_async0();
    __syncthreads();

    const int NK = IN_F / BK;             // 128 K-steps
    unsigned cur = 0;                      // element offset toggle: 0 <-> BUFSZ
    #pragma unroll 1
    for (int kb = 0; kb < NK; ++kb) {
        // ---- issue async loads for next tile into other buffer ----
        if (kb + 1 < NK) {
            const unsigned nxt = cur ^ BUFSZ;
            const int kBase = (kb + 1) * BK;
            async_copy16(A + (size_t)(mBase0 + r0     ) * IN_F + kBase + q * 8,
                         &AshF[nxt + stA0]);
            async_copy16(A + (size_t)(mBase0 + r0 + 64) * IN_F + kBase + q * 8,
                         &AshF[nxt + stA1]);
            async_copy16(B + (size_t)(nBase0 + r0     ) * IN_F + kBase + q * 8,
                         &BshF[nxt + stA0]);
            async_copy16(B + (size_t)(nBase0 + r0 + 64) * IN_F + kBase + q * 8,
                         &BshF[nxt + stA1]);
        }

        // ---- fragments per ISA 7.12.2 wave32 layouts, as 16B LDS loads ----
        // A 16x32: lane m=lane%16, half=lane/16:
        //   u[0..3] = K pairs half*8+{0,2,4,6}   -> 16B quad at index half
        //   u[4..7] = K pairs 16+half*8+{0,2,4,6}-> 16B quad at index 2+half
        FragBF a[2], b[4];
        #pragma unroll
        for (int i = 0; i < 2; ++i) {
            const v4i_vs* rowq = (const v4i_vs*)&AshF[cur + aRow[i]];
            a[i].q[0] = rowq[half];
            a[i].q[1] = rowq[2 + half];
        }
        // B 32x16: lane n=lane%16, kh=lane/16 (same K order as A):
        //   u[0..3] = K kh*8+{0..7} -> quad kh; u[4..7] = K 16+kh*8+{0..7} -> quad 2+kh
        #pragma unroll
        for (int i = 0; i < 4; ++i) {
            const v4i_vs* rowq = (const v4i_vs*)&BshF[cur + bRow[i]];
            b[i].q[0] = rowq[half];
            b[i].q[1] = rowq[2 + half];
        }

        // ---- 8 WMMAs: D = A x B + C ----
        #pragma unroll
        for (int i = 0; i < 2; ++i)
            #pragma unroll
            for (int j = 0; j < 4; ++j)
                acc[i][j] = __builtin_amdgcn_wmma_f32_16x16x32_bf16(
                    false, a[i].v, false, b[j].v,
                    (short)0, acc[i][j], false, false);

        // ---- single sync point: my async writes done + everyone done reading
        wait_async0();
        __syncthreads();
        cur ^= BUFSZ;
    }

    // ---- epilogue: C/D layout: lane col n=lane%16, VGPR r -> row r+8*half
    #pragma unroll
    for (int i = 0; i < 2; ++i)
        #pragma unroll
        for (int j = 0; j < 4; ++j)
            #pragma unroll
            for (int r = 0; r < 8; ++r) {
                int m = mBase0 + waveM * 32 + i * 16 + half * 8 + r;
                int n = nBase0 + waveN * 64 + j * 16 + m16;
                C[(size_t)m * OUT_F + n] = acc[i][j][r];
            }
}

// ---------------- Kernel 4: y = fwht(y)*SV + bias (in place) --------------
__global__ void __launch_bounds__(THREADS)
k_had_out(float* __restrict__ y, const float* __restrict__ SV,
          const float* __restrict__ bias) {
    __shared__ float buf[OUT_F];
    const int row = blockIdx.x;
    const int tid = threadIdx.x;
    float* yr = y + (size_t)row * OUT_F;
    #pragma unroll
    for (int t = 0; t < 16; ++t) {
        int i = tid + t * THREADS;
        buf[i] = yr[i];
    }
    __syncthreads();
    fwht4096(buf, tid);
    const float inv = 1.0f / 64.0f;
    #pragma unroll
    for (int t = 0; t < 16; ++t) {
        int i = tid + t * THREADS;
        yr[i] = buf[i] * inv * SV[i] + bias[i];
    }
}

// ---------------------------------------------------------------------------
extern "C" void kernel_launch(void* const* d_in, const int* in_sizes, int n_in,
                              void* d_out, int out_size, void* d_ws, size_t ws_size,
                              hipStream_t stream) {
    const float* x     = (const float*)d_in[0];
    const int*   Qidxs = (const int*)  d_in[1];
    const float* cbk   = (const float*)d_in[2];
    const float* SU    = (const float*)d_in[3];
    const float* SV    = (const float*)d_in[4];
    const float* bias  = (const float*)d_in[5];
    float*       out   = (float*)d_out;

    __bf16* hbuf = (__bf16*)d_ws;                                     // 64 MB
    __bf16* Wbuf = (__bf16*)((char*)d_ws + (size_t)NROWS * IN_F * 2); // +32 MB

    k_had_in <<<NROWS, THREADS, 0, stream>>>(x, SU, hbuf);
    k_decode <<<(OUT_F * (IN_F / 8)) / THREADS, THREADS, 0, stream>>>(Qidxs, cbk, Wbuf);
    k_gemm   <<<dim3(OUT_F / BN, NROWS / BM), THREADS, 0, stream>>>(hbuf, Wbuf, out);
    k_had_out<<<NROWS, THREADS, 0, stream>>>(out, SV, bias);
}
